// BidirectionalLSTMWithAttention_9328668967819
// MI455X (gfx1250) — matrix operand order repaired and checked by reading, not compiled, not run on hardware
//
#include <hip/hip_runtime.h>
#include <hip/hip_bf16.h>

// ---------------------------------------------------------------------------
// BidirectionalLSTMWithAttention for MI455X (gfx1250), wave32 + WMMA bf16.
//
//   1) convert weights fp32->bf16 into ws (Whh [512][128], Wih1 [512][256],
//      Wih0 zero-padded to [512][32]); combine biases bih+bhh.
//   2) lstm_kernel<0>: grid(32 batch tiles, 2 dirs). Whh + Wih0 B-fragments
//      held in persistent VGPRs (time-invariant), h/c state in LDS (h in
//      natural k-order => one 32B ds read per A fragment),
//      per-step gates via v_wmma_f32_16x16x32_bf16; branchless wave-uniform
//      activations (tanh(v) = 2*sigmoid(2v)-1) via v_exp_f32 + v_rcp_f32.
//      Writes h0cat bf16 [B][T][256] (fwd cols 0..127, bwd cols 128..255).
//   3) lstm_kernel<1>: Whh fragments in VGPRs, Wih (256KB) staged in LDS,
//      input = h0cat (K=256), writes h1cat.
//   4) attn_kernel: per-batch softmax over T + weighted sum -> context [B][256].
//   5) head_kernel: fc1/res/fc2/out -> d_out [B][6] fp32.
// ---------------------------------------------------------------------------

typedef __bf16 bf16_t;
typedef __attribute__((ext_vector_type(16))) __bf16 v16bf;
typedef __attribute__((ext_vector_type(8)))  __bf16 v8bf;
typedef __attribute__((ext_vector_type(2)))  __bf16 v2bf;
typedef __attribute__((ext_vector_type(8)))  float   v8f;

#define BATCH 512
#define TLEN  512
#define HID   128
#define GATES 512   // 4*HID
#define D0    13
#define K0P   32    // layer0 input K padded to one 32-deep WMMA slice
#define D1    256   // layer1 input dim

static __device__ inline v8f zero8() {
    v8f z;
#pragma unroll
    for (int i = 0; i < 8; ++i) z[i] = 0.f;
    return z;
}

static __device__ inline v8f wmma_bf16(v16bf a, v16bf b, v8f c) {
    // D = A(16x32 bf16) * B(32x16 bf16) + C(16x16 f32)
    return __builtin_amdgcn_wmma_f32_16x16x32_bf16(
        /*neg_a=*/false, a, /*neg_b=*/false, b,
        /*c_mod=*/(short)0, c, /*reuse_a=*/false, /*reuse_b=*/false);
}

static __device__ inline v16bf cat8(v8bf lo, v8bf hi) {
    return __builtin_shufflevector(lo, hi, 0,1,2,3,4,5,6,7,8,9,10,11,12,13,14,15);
}

// Fast activations: v_exp_f32 + v_rcp_f32, no precise-division expansion.
static __device__ inline float sigmoid_fast(float v) {
    return __builtin_amdgcn_rcpf(1.f + __expf(-v));
}
static __device__ inline float tanh_fast(float v) {
    // tanh(v) = 1 - 2/(exp(2v)+1); saturates to +/-1 correctly at large |v|.
    return 1.f - 2.f * __builtin_amdgcn_rcpf(__expf(2.f * v) + 1.f);
}

// Permute k within each 32-block: natural order -> [0..7, 16..23, 8..15, 24..31]
// so that lane-group kg reads one contiguous v16bf per A fragment.
static __device__ inline int permk(int j) {
    const int r8  = (j >> 3) & 3;
    const int r8p = (r8 == 1) ? 2 : ((r8 == 2) ? 1 : r8);
    return (j & ~31) | (r8p << 3) | (j & 7);
}

// ---------------------------------------------------------------------------
// LSTM layer kernel. One WG = 16 batch rows x all 512 gate columns.
// grid.x = BATCH/16 = 32, grid.y = 2 (0=fwd, 1=bwd). 256 threads = 8 waves.
// Wave w owns gate columns [w*64, w*64+64) => gate id = w>>1 (i,f,g,o).
// ---------------------------------------------------------------------------
template <int LAYER>
__global__ __launch_bounds__(256)
void lstm_kernel(const float*  __restrict__ x,     // LAYER==0: [B][T*13] fp32
                 const bf16_t* __restrict__ hin,   // LAYER==1: [B][T][256] bf16
                 const bf16_t* __restrict__ WihF, const bf16_t* __restrict__ WihB,
                 const bf16_t* __restrict__ WhhF, const bf16_t* __restrict__ WhhB,
                 const float*  __restrict__ biasF, const float* __restrict__ biasB,
                 bf16_t*       __restrict__ hout)  // [B][T][256], dir selects 128-col half
{
    const int dir  = blockIdx.y;
    const int b0   = blockIdx.x * 16;
    const int tid  = threadIdx.x;
    const int wave = tid >> 5;
    const int lane = tid & 31;
    const int nl   = lane & 15;   // A: row m   | B: col n | C/D: col n
    const int kg   = lane >> 4;   // lane K-group

    const bf16_t* Wih  = dir ? WihB  : WihF;
    const bf16_t* Whh  = dir ? WhhB  : WhhF;
    const float*  bias = dir ? biasB : biasF;

    extern __shared__ char smem[];
    char*   p    = smem;
    bf16_t* sWih = nullptr;
    if (LAYER == 1) { sWih = (bf16_t*)p; p += (size_t)GATES * D1 * sizeof(bf16_t); } // 256 KB
    float*  sGates = (float*)p;  p += 4 * 16 * HID * sizeof(float); // [gate][m][j]
    float*  sC     = (float*)p;  p += 16 * HID * sizeof(float);     // [16][128]
    bf16_t* sH     = (bf16_t*)p;                                    // [16][128] (natural k)

    // --- persistent register-resident B fragments (time-invariant) ---
    v16bf bh[4][4];                        // [ks][nt] recurrent Whh fragments
#pragma unroll
    for (int ks = 0; ks < 4; ++ks)
#pragma unroll
        for (int nt = 0; nt < 4; ++nt)
            bh[ks][nt] = *(const v16bf*)(Whh + (size_t)(wave * 64 + nt * 16 + nl) * HID
                                             + ks * 32 + kg * 16);
    v16bf bx[4];                           // layer0 input Wih fragments
    if (LAYER == 0) {
#pragma unroll
        for (int nt = 0; nt < 4; ++nt)
            bx[nt] = *(const v16bf*)(Wih + (size_t)(wave * 64 + nt * 16 + nl) * K0P + kg * 16);
    }

    // --- stage layer1 input weights into LDS, init state ---
    if (LAYER == 1)
        for (int i = tid * 8; i < GATES * D1; i += 256 * 8)
            *(v8bf*)(sWih + i) = *(const v8bf*)(Wih + i);
    for (int i = tid; i < 16 * HID; i += 256) { sC[i] = 0.f; sH[i] = (bf16_t)0.f; }
    __syncthreads();

    const int gate = wave >> 1;
    // Branchless wave-uniform activation: tanh(v) = 2*sigmoid(2v) - 1.
    // gate 2 (g) uses tanh; gates 0,1,3 (i,f,o) use sigmoid.
    const float aIn  = (gate == 2) ? 2.f : 1.f;
    const float aOut = (gate == 2) ? 2.f : 1.f;
    const float aOff = (gate == 2) ? -1.f : 0.f;

    float bcol[4];
#pragma unroll
    for (int nt = 0; nt < 4; ++nt) bcol[nt] = bias[wave * 64 + nt * 16 + nl];

    for (int step = 0; step < TLEN; ++step) {
        const int t = dir ? (TLEN - 1 - step) : step;

        v8f acc[4];
#pragma unroll
        for (int nt = 0; nt < 4; ++nt) acc[nt] = zero8();

        // ---------------- input projection x_t @ Wih.T ----------------
        if (LAYER == 0) {
            // A fragment from x (K padded 13 -> 32 with zeros)
            const float* xb = x + (size_t)(b0 + nl) * (TLEN * D0) + (size_t)t * D0;
            const int kb = kg * 16;
            v16bf ax;
#pragma unroll
            for (int e = 0; e < 8; ++e) {
                const int k = kb + e;
                ax[e]     = (bf16_t)((k < D0) ? xb[k] : 0.f);
                ax[e + 8] = (bf16_t)((k + 8 < D0) ? xb[k + 8] : 0.f);          // k = kb+8..kb+15, zero padded past 13
            }
#pragma unroll
            for (int nt = 0; nt < 4; ++nt)
                acc[nt] = wmma_bf16(ax, bx[nt], acc[nt]);
        } else {
            const bf16_t* hrow = hin + ((size_t)(b0 + nl) * TLEN + t) * D1;
            {   // prefetch next step's input row (global_prefetch_b8)
                const int tn = dir ? (t > 0 ? t - 1 : 0) : (t < TLEN - 1 ? t + 1 : t);
                __builtin_prefetch(hin + ((size_t)(b0 + nl) * TLEN + tn) * D1, 0, 1);
            }
#pragma unroll
            for (int ks = 0; ks < 8; ++ks) {          // K = 256 = 8 x 32
                const int kb = ks * 32 + kg * 16;
                const v8bf lo = *(const v8bf*)(hrow + kb);
                const v8bf hi = *(const v8bf*)(hrow + kb + 8);
                const v16bf ai = cat8(lo, hi);
#pragma unroll
                for (int nt = 0; nt < 4; ++nt) {
                    const int g = wave * 64 + nt * 16 + nl;
                    const v16bf bi = *(const v16bf*)(sWih + (size_t)g * D1 + ks * 32 + kg * 16);
                    acc[nt] = wmma_bf16(ai, bi, acc[nt]);
                }
            }
        }

        // ------- recurrent projection h @ Whh.T (B fragments in VGPRs) -------
#pragma unroll
        for (int ks = 0; ks < 4; ++ks) {              // K = 128 = 4 x 32
            const v16bf ah = *(const v16bf*)(sH + nl * HID + ks * 32 + kg * 16);
#pragma unroll
            for (int nt = 0; nt < 4; ++nt)
                acc[nt] = wmma_bf16(ah, bh[ks][nt], acc[nt]);
        }

        // -------- bias + gate nonlinearity, scatter to LDS gate buffer --------
        // C/D layout: VGPR r, lanes0-15 -> (m=r, n=lane); lanes16-31 -> (m=r+8).
#pragma unroll
        for (int nt = 0; nt < 4; ++nt) {
            const int j = (wave & 1) * 64 + nt * 16 + nl;   // column within gate
#pragma unroll
            for (int r = 0; r < 8; ++r) {
                const float v = acc[nt][r] + bcol[nt];
                const float a = aOut * sigmoid_fast(aIn * v) + aOff;  // branchless
                const int m = r + 8 * kg;
                sGates[(gate * 16 + m) * HID + j] = a;
            }
        }
        __syncthreads();

        // -------- c/h update (j-pairs): c = f*c + i*g ; h = o*tanh(c) --------
        for (int i = tid; i < 8 * HID; i += 256) {     // 1024 column pairs
            const int m  = i >> 6;
            const int jp = (i & 63) << 1;              // j = jp, jp+1
            const float2 ig = *(const float2*)(sGates + (0 * 16 + m) * HID + jp);
            const float2 fg = *(const float2*)(sGates + (1 * 16 + m) * HID + jp);
            const float2 gg = *(const float2*)(sGates + (2 * 16 + m) * HID + jp);
            const float2 og = *(const float2*)(sGates + (3 * 16 + m) * HID + jp);
            float2* cp = (float2*)(sC + m * HID + jp);
            float2 c = *cp;
            c.x = fg.x * c.x + ig.x * gg.x;
            c.y = fg.y * c.y + ig.y * gg.y;
            *cp = c;
            v2bf hp;
            hp[0] = (bf16_t)(og.x * tanh_fast(c.x));
            hp[1] = (bf16_t)(og.y * tanh_fast(c.y));
            // pairs stay inside one 8-chunk => permk(jp+1) == permk(jp)+1
            *(v2bf*)(sH + m * HID + jp) = hp;                 // natural k order
            *(v2bf*)(hout + ((size_t)(b0 + m) * TLEN + t) * 256 + dir * 128 + jp) = hp;
        }
        __syncthreads();
    }
}

// ---------------------------------------------------------------------------
// Attention: scores = h1 @ attn_w.T + b, softmax over T, context = sum w*h1.
// One WG (256 threads) per batch row.
// ---------------------------------------------------------------------------
__global__ __launch_bounds__(256)
void attn_kernel(const bf16_t* __restrict__ h1, const float* __restrict__ attn_w,
                 const float* __restrict__ attn_b, float* __restrict__ ctx)
{
    __shared__ float sS[TLEN];
    __shared__ float sRed[256];
    __shared__ float sW[256];
    const int b = blockIdx.x, tid = threadIdx.x;

    sW[tid] = attn_w[tid];
    __syncthreads();

    for (int t = tid; t < TLEN; t += 256) {
        const bf16_t* hp = h1 + ((size_t)b * TLEN + t) * 256;
        float s = 0.f;
        for (int k = 0; k < 256; k += 8) {
            const v8bf hv = *(const v8bf*)(hp + k);
#pragma unroll
            for (int e = 0; e < 8; ++e) s += (float)hv[e] * sW[k + e];
        }
        sS[t] = s + attn_b[0];
    }
    __syncthreads();

    float m = -1e30f;
    for (int t = tid; t < TLEN; t += 256) m = fmaxf(m, sS[t]);
    sRed[tid] = m; __syncthreads();
    for (int off = 128; off > 0; off >>= 1) {
        if (tid < off) sRed[tid] = fmaxf(sRed[tid], sRed[tid + off]);
        __syncthreads();
    }
    m = sRed[0];
    __syncthreads();

    float sum = 0.f;
    for (int t = tid; t < TLEN; t += 256) {
        const float e = __expf(sS[t] - m);
        sS[t] = e;
        sum += e;
    }
    sRed[tid] = sum; __syncthreads();
    for (int off = 128; off > 0; off >>= 1) {
        if (tid < off) sRed[tid] += sRed[tid + off];
        __syncthreads();
    }
    const float inv = __builtin_amdgcn_rcpf(sRed[0]);

    float acc = 0.f;                                  // thread owns feature k = tid
    for (int t = 0; t < TLEN; ++t)
        acc += sS[t] * (float)h1[((size_t)b * TLEN + t) * 256 + tid];
    ctx[(size_t)b * 256 + tid] = acc * inv;
}

// ---------------------------------------------------------------------------
// FC head: fc1=relu(ctx@fc1_w.T+b); res=ctx@res_w.T+b;
//          fc2=relu((fc1+res)@fc2_w.T+b); out=fc2@out_w.T+b.  One WG per row.
// ---------------------------------------------------------------------------
__global__ __launch_bounds__(64)
void head_kernel(const float* __restrict__ ctx,
                 const float* __restrict__ fc1_w, const float* __restrict__ fc1_b,
                 const float* __restrict__ res_w, const float* __restrict__ res_b,
                 const float* __restrict__ fc2_w, const float* __restrict__ fc2_b,
                 const float* __restrict__ out_w, const float* __restrict__ out_b,
                 float* __restrict__ out)
{
    __shared__ float sCtx[256];
    __shared__ float sS[64];
    const int b = blockIdx.x, tid = threadIdx.x;

    for (int k = tid; k < 256; k += 64) sCtx[k] = ctx[(size_t)b * 256 + k];
    __syncthreads();

    float a1 = fc1_b[tid], ar = res_b[tid];
    for (int k = 0; k < 256; ++k) {
        const float c = sCtx[k];
        a1 += c * fc1_w[tid * 256 + k];
        ar += c * res_w[tid * 256 + k];
    }
    a1 = fmaxf(a1, 0.f);
    sS[tid] = a1 + ar;
    __syncthreads();

    float a2 = fc2_b[tid];
    for (int k = 0; k < 64; ++k) a2 += sS[k] * fc2_w[tid * 64 + k];
    a2 = fmaxf(a2, 0.f);
    __syncthreads();
    sCtx[tid] = a2;           // reuse as fc2 buffer
    __syncthreads();

    if (tid < 6) {
        float o = out_b[tid];
        for (int k = 0; k < 64; ++k) o += sCtx[k] * out_w[tid * 64 + k];
        out[(size_t)b * 6 + tid] = o;
    }
}

// ---------------------------------------------------------------------------
// Weight prep kernels.
// ---------------------------------------------------------------------------
__global__ void cvt_bf16_kernel(const float* __restrict__ in, bf16_t* __restrict__ out, int n) {
    const int i = blockIdx.x * 256 + threadIdx.x;
    if (i < n) out[i] = (bf16_t)in[i];
}

__global__ void pack_wih0_kernel(const float* __restrict__ in, bf16_t* __restrict__ out) {
    const int i = blockIdx.x * 256 + threadIdx.x;        // over 512*32
    if (i < GATES * K0P) {
        const int g = i >> 5, k = i & 31;
        out[i] = (bf16_t)((k < D0) ? in[g * D0 + k] : 0.f);
    }
}

__global__ void bias_kernel(const float* __restrict__ bih, const float* __restrict__ bhh,
                            float* __restrict__ out) {
    const int i = blockIdx.x * 256 + threadIdx.x;
    if (i < GATES) out[i] = bih[i] + bhh[i];
}

// ---------------------------------------------------------------------------
extern "C" void kernel_launch(void* const* d_in, const int* in_sizes, int n_in,
                              void* d_out, int out_size, void* d_ws, size_t ws_size,
                              hipStream_t stream)
{
    (void)in_sizes; (void)n_in; (void)out_size; (void)ws_size;

    const float* x      = (const float*)d_in[0];
    const float* Wih0f  = (const float*)d_in[1];
    const float* Whh0f  = (const float*)d_in[2];
    const float* bih0f  = (const float*)d_in[3];
    const float* bhh0f  = (const float*)d_in[4];
    const float* Wih0b  = (const float*)d_in[5];
    const float* Whh0b  = (const float*)d_in[6];
    const float* bih0b  = (const float*)d_in[7];
    const float* bhh0b  = (const float*)d_in[8];
    const float* Wih1f  = (const float*)d_in[9];
    const float* Whh1f  = (const float*)d_in[10];
    const float* bih1f  = (const float*)d_in[11];
    const float* bhh1f  = (const float*)d_in[12];
    const float* Wih1b  = (const float*)d_in[13];
    const float* Whh1b  = (const float*)d_in[14];
    const float* bih1b  = (const float*)d_in[15];
    const float* bhh1b  = (const float*)d_in[16];
    const float* attn_w = (const float*)d_in[17];
    const float* attn_b = (const float*)d_in[18];
    const float* res_w  = (const float*)d_in[19];
    const float* res_b  = (const float*)d_in[20];
    const float* fc1_w  = (const float*)d_in[21];
    const float* fc1_b  = (const float*)d_in[22];
    const float* fc2_w  = (const float*)d_in[23];
    const float* fc2_b  = (const float*)d_in[24];
    const float* out_w  = (const float*)d_in[25];
    const float* out_b  = (const float*)d_in[26];

    char* ws = (char*)d_ws;
    size_t off = 0;
    auto alloc = [&](size_t bytes) -> void* {
        void* p = ws + off;
        off += (bytes + 255) & ~(size_t)255;
        return p;
    };

    bf16_t* wWhh0f = (bf16_t*)alloc((size_t)GATES * HID * 2);
    bf16_t* wWhh0b = (bf16_t*)alloc((size_t)GATES * HID * 2);
    bf16_t* wWhh1f = (bf16_t*)alloc((size_t)GATES * HID * 2);
    bf16_t* wWhh1b = (bf16_t*)alloc((size_t)GATES * HID * 2);
    bf16_t* wWih0f = (bf16_t*)alloc((size_t)GATES * K0P * 2);
    bf16_t* wWih0b = (bf16_t*)alloc((size_t)GATES * K0P * 2);
    bf16_t* wWih1f = (bf16_t*)alloc((size_t)GATES * D1 * 2);
    bf16_t* wWih1b = (bf16_t*)alloc((size_t)GATES * D1 * 2);
    float*  bias0f = (float*)alloc((size_t)GATES * 4);
    float*  bias0b = (float*)alloc((size_t)GATES * 4);
    float*  bias1f = (float*)alloc((size_t)GATES * 4);
    float*  bias1b = (float*)alloc((size_t)GATES * 4);
    bf16_t* h0     = (bf16_t*)alloc((size_t)BATCH * TLEN * 256 * 2);  // 128 MB
    bf16_t* h1     = (bf16_t*)alloc((size_t)BATCH * TLEN * 256 * 2);  // 128 MB
    float*  ctx    = (float*)alloc((size_t)BATCH * 256 * 4);

    // --- weight prep ---
    const int nWhh = GATES * HID, nWih1 = GATES * D1, nWih0 = GATES * K0P;
    cvt_bf16_kernel<<<(nWhh + 255) / 256, 256, 0, stream>>>(Whh0f, wWhh0f, nWhh);
    cvt_bf16_kernel<<<(nWhh + 255) / 256, 256, 0, stream>>>(Whh0b, wWhh0b, nWhh);
    cvt_bf16_kernel<<<(nWhh + 255) / 256, 256, 0, stream>>>(Whh1f, wWhh1f, nWhh);
    cvt_bf16_kernel<<<(nWhh + 255) / 256, 256, 0, stream>>>(Whh1b, wWhh1b, nWhh);
    cvt_bf16_kernel<<<(nWih1 + 255) / 256, 256, 0, stream>>>(Wih1f, wWih1f, nWih1);
    cvt_bf16_kernel<<<(nWih1 + 255) / 256, 256, 0, stream>>>(Wih1b, wWih1b, nWih1);
    pack_wih0_kernel<<<(nWih0 + 255) / 256, 256, 0, stream>>>(Wih0f, wWih0f);
    pack_wih0_kernel<<<(nWih0 + 255) / 256, 256, 0, stream>>>(Wih0b, wWih0b);
    bias_kernel<<<2, 256, 0, stream>>>(bih0f, bhh0f, bias0f);
    bias_kernel<<<2, 256, 0, stream>>>(bih0b, bhh0b, bias0b);
    bias_kernel<<<2, 256, 0, stream>>>(bih1f, bhh1f, bias1f);
    bias_kernel<<<2, 256, 0, stream>>>(bih1b, bhh1b, bias1b);

    // --- LSTM layers (fwd+bwd concurrent via grid.y) ---
    const size_t ldsState = (size_t)4 * 16 * HID * 4 + 16 * HID * 4 + 16 * HID * 2; // 44 KB
    const size_t lds0 = ldsState;                                   // ~44 KB
    const size_t lds1 = (size_t)GATES * D1 * 2 + ldsState;          // ~300 KB

    lstm_kernel<0><<<dim3(BATCH / 16, 2, 1), 256, lds0, stream>>>(
        x, (const bf16_t*)nullptr, wWih0f, wWih0b, wWhh0f, wWhh0b, bias0f, bias0b, h0);
    lstm_kernel<1><<<dim3(BATCH / 16, 2, 1), 256, lds1, stream>>>(
        (const float*)nullptr, h0, wWih1f, wWih1b, wWhh1f, wWhh1b, bias1f, bias1b, h1);

    // --- attention + head ---
    attn_kernel<<<BATCH, 256, 0, stream>>>(h1, attn_w, attn_b, ctx);
    head_kernel<<<BATCH, 64, 0, stream>>>(ctx, fc1_w, fc1_b, res_w, res_b,
                                          fc2_w, fc2_b, out_w, out_b, (float*)d_out);
}
